// Model_39676907887714
// MI455X (gfx1250) — matrix operand order repaired and checked by reading, not compiled, not run on hardware
//
#include <hip/hip_runtime.h>
#include <hip/hip_bf16.h>

typedef __attribute__((ext_vector_type(16))) _Float16 v16h;
typedef __attribute__((ext_vector_type(8)))  _Float16 v8h;
typedef __attribute__((ext_vector_type(8)))  float    v8f;

#define B_   8
#define S_   4096
#define DIN_ 25
#define D_   10
#define DP_  16   // head dim padded to 16 (WMMA K uses 32 with zeros in 16..31)

#define LOG2E_ 1.4426950408889634f

union H16u { unsigned short u; _Float16 f; };
__device__ inline _Float16 ush_as_h(unsigned v) { H16u c; c.u = (unsigned short)v; return c.f; }

__device__ inline float fast_exp2(float x) {
#if __has_builtin(__builtin_amdgcn_exp2f)
    return __builtin_amdgcn_exp2f(x);   // v_exp_f32
#else
    return exp2f(x);
#endif
}

__device__ inline unsigned pack_h2(float a, float b) {
#if __has_builtin(__builtin_amdgcn_cvt_pkrtz)
    auto pk = __builtin_amdgcn_cvt_pkrtz(a, b);      // v_cvt_pk_rtz_f16_f32
    return __builtin_bit_cast(unsigned, pk);
#else
    H16u lo, hi; lo.f = (_Float16)a; hi.f = (_Float16)b;
    return (unsigned)lo.u | ((unsigned)hi.u << 16);
#endif
}

// ---------------------------------------------------------------------------
// Kernel 1: fused projections.
//   v3 = (x1+x2) @ W_lin^T + 2*b_lin;  k/q/v = v3 @ W{k,q,v}^T + b{k,q,v}
// K is pre-scaled by log2(e) so attention scores land in the exp2 domain.
// Writes f16: Kp[row][16], Qp[row][16] (d>=10 zero), Vt[b][d][s] transposed.
// ---------------------------------------------------------------------------
__global__ __launch_bounds__(256) void proj_kernel(
    const float* __restrict__ x1, const float* __restrict__ x2,
    const float* __restrict__ W_lin, const float* __restrict__ b_lin,
    const float* __restrict__ Wk, const float* __restrict__ bk,
    const float* __restrict__ Wq, const float* __restrict__ bq,
    const float* __restrict__ Wv, const float* __restrict__ bv,
    _Float16* __restrict__ Kp, _Float16* __restrict__ Qp,
    _Float16* __restrict__ Vt)
{
    const int row = blockIdx.x * blockDim.x + threadIdx.x;   // 0 .. B*S-1
    if (row >= B_ * S_) return;

    float xs[DIN_];
#pragma unroll
    for (int i = 0; i < DIN_; ++i)
        xs[i] = x1[(size_t)row * DIN_ + i] + x2[(size_t)row * DIN_ + i];

    float v3[D_];
#pragma unroll
    for (int d = 0; d < D_; ++d) {
        float a = 2.0f * b_lin[d];
#pragma unroll
        for (int i = 0; i < DIN_; ++i)
            a = fmaf(W_lin[d * DIN_ + i], xs[i], a);
        v3[d] = a;
    }

    const int b = row / S_;
    const int s = row % S_;

#pragma unroll
    for (int d = 0; d < DP_; ++d) {
        float kd = 0.f, qd = 0.f, vd = 0.f;
        if (d < D_) {
            kd = bk[d]; qd = bq[d]; vd = bv[d];
#pragma unroll
            for (int j = 0; j < D_; ++j) {
                kd = fmaf(Wk[d * D_ + j], v3[j], kd);
                qd = fmaf(Wq[d * D_ + j], v3[j], qd);
                vd = fmaf(Wv[d * D_ + j], v3[j], vd);
            }
            kd *= LOG2E_;   // exp2-domain softmax downstream
        }
        Kp[(size_t)row * DP_ + d] = (_Float16)kd;
        Qp[(size_t)row * DP_ + d] = (_Float16)qd;
        Vt[((size_t)b * DP_ + d) * S_ + s] = (_Float16)vd;
    }
}

// ---------------------------------------------------------------------------
// Kernel 2: flash attention, one wave per 16 s-rows of one batch.
//   scores[s][t] = k[s].q[t]  (softmax over t),  out[s] = sum_t attn * v[t]
// Computed transposed: C'[t][s] = q[t].k[s]; O^T[d][s] = V^T P^T, K-dim = 32 t.
// Softmax runs in exp2 domain (K pre-scaled by log2 e).
// ---------------------------------------------------------------------------
__global__ __launch_bounds__(256) void attn_kernel(
    const _Float16* __restrict__ Kp, const _Float16* __restrict__ Qp,
    const _Float16* __restrict__ Vt, float* __restrict__ out)
{
    const int lane = threadIdx.x & 31;
    const int n    = lane & 15;   // N of C tiles (= s in tile); also M for A operands
    const int h    = lane >> 4;   // lane group
    const int wave = blockIdx.x * (blockDim.x >> 5) + (threadIdx.x >> 5);
    const int tilesPerB = S_ / 16;
    const int b  = wave / tilesPerB;
    const int s0 = (wave % tilesPerB) * 16;

    // ---- B1 operand: this wave's 16 K-rows as 32x16 (K-dim = padded head dim).
    // B layout: group g halves 0..7 -> K=8g+e (d=8g+e); halves 8..15 -> K=16..31 (zeros).
    v16h zero16;
#pragma unroll
    for (int i = 0; i < 16; ++i) zero16[i] = (_Float16)0.0f;

    const v8h krow = *(const v8h*)(Kp + ((size_t)b * S_ + s0 + n) * DP_ + h * 8);
    v16h b1 = zero16;
#pragma unroll
    for (int i = 0; i < 8; ++i) b1[i] = krow[i];

    v8f acc = {};                 // O^T accumulator: [d = r + 8h][s = n]
    float m = -1e30f;             // shared (synced) running max, exp2 domain
    float l = 0.0f;               // PARTIAL row sum (own t-subset only)

    const v8f czero = {};
    const _Float16* __restrict__ vbase = Vt + ((size_t)b * DP_ + n) * S_;
    const _Float16* __restrict__ qbase = Qp + ((size_t)b * S_) * DP_;

    for (int t0 = 0; t0 < S_; t0 += 32) {
        __builtin_prefetch(qbase + (size_t)(t0 + 32 + n) * DP_, 0, 1);
        __builtin_prefetch(vbase + t0 + 32, 0, 1);

        // ---- First GEMMs: C'[t][s] for t-tiles [t0,t0+16) and [t0+16,t0+32).
        // A layout (16-bit 16x32): group 0 halves 0..7 -> K=0..7 (16..23 zero);
        // group 1 halves 0..7 -> K=8..15. One 16B load + zero upper halves.
        v16h a0 = zero16, a1 = zero16;
        {
            const v8h q0 = *(const v8h*)(qbase + (size_t)(t0 + n) * DP_ + h * 8);
            const v8h q1 = *(const v8h*)(qbase + (size_t)(t0 + 16 + n) * DP_ + h * 8);
#pragma unroll
            for (int i = 0; i < 8; ++i) { a0[i] = q0[i]; a1[i] = q1[i]; }
        }
        v8f c0 = __builtin_amdgcn_wmma_f32_16x16x32_f16(false, a0, false, b1,
                                                        (short)0, czero, false, false);
        v8f c1 = __builtin_amdgcn_wmma_f32_16x16x32_f16(false, a1, false, b1,
                                                        (short)0, czero, false, false);

        // ---- Online softmax over the 32 t values (fixed s = lane), exp2 domain.
        float tmax = c0[0];
#pragma unroll
        for (int r = 1; r < 8; ++r) tmax = fmaxf(tmax, c0[r]);
#pragma unroll
        for (int r = 0; r < 8; ++r) tmax = fmaxf(tmax, c1[r]);
        tmax = fmaxf(tmax, __shfl_xor(tmax, 16));      // sync max across lane groups

        const float mnew  = fmaxf(m, tmax);
        const float alpha = fast_exp2(m - mnew);

        float p0[8], p1[8], psum = 0.0f;
#pragma unroll
        for (int r = 0; r < 8; ++r) {
            p0[r] = fast_exp2(c0[r] - mnew); psum += p0[r];
            p1[r] = fast_exp2(c1[r] - mnew); psum += p1[r];
        }
        l = l * alpha + psum;      // partial sum; combined across groups at the end
        m = mnew;
#pragma unroll
        for (int r = 0; r < 8; ++r) acc[r] *= alpha;

        // ---- Build B2 = P^T (32x16, K = t_local): C-layout -> B-layout with
        // packed-f16 xor16 shuffles (one dword carries tile0+tile1 for a given r).
        v16h b2;
#pragma unroll
        for (int r = 0; r < 8; ++r) {
            const unsigned own = pack_h2(p0[r], p1[r]);          // {p0h, p1h}
            const unsigned swp = (unsigned)__shfl_xor((int)own, 16);
            const unsigned lo  = h ? (swp >> 16) : (own & 0xffffu); // K=r
            const unsigned hi  = h ? (own >> 16) : (swp & 0xffffu); // K=8+r
            b2[r]     = ush_as_h(lo);
            b2[8 + r] = ush_as_h(hi);
        }

        // ---- A2 = V^T (16 x 32): M = d = n, K = t_local; contiguous f16 loads.
        const v8h va = *(const v8h*)(vbase + t0 + h * 16);       // t_local = h*16 + 0..7
        const v8h vb = *(const v8h*)(vbase + t0 + h * 16 + 8);   // t_local = h*16 + 8..15
        v16h a2;
#pragma unroll
        for (int i = 0; i < 8; ++i) { a2[i] = va[i]; a2[8 + i] = vb[i]; }

        acc = __builtin_amdgcn_wmma_f32_16x16x32_f16(false, a2, false, b2,
                                                     (short)0, acc, false, false);
    }

    // ---- Epilogue: combine partial sums, out[b][s0+n][d = r + 8h] = acc[r]/l.
    const float ltot = l + __shfl_xor(l, 16);
    const float inv  = 1.0f / ltot;
    float* __restrict__ orow = out + ((size_t)b * S_ + s0 + n) * D_;
    if (h == 0) {
#pragma unroll
        for (int r = 0; r < 8; r += 2) {
            float2 vv; vv.x = acc[r] * inv; vv.y = acc[r + 1] * inv;
            *(float2*)(orow + r) = vv;     // 8-byte aligned (row stride 40B)
        }
    } else {
        float2 vv; vv.x = acc[0] * inv; vv.y = acc[1] * inv;   // d = 8, 9
        *(float2*)(orow + 8) = vv;
    }
}

// ---------------------------------------------------------------------------
extern "C" void kernel_launch(void* const* d_in, const int* in_sizes, int n_in,
                              void* d_out, int out_size, void* d_ws, size_t ws_size,
                              hipStream_t stream) {
    const float* x1    = (const float*)d_in[0];
    const float* x2    = (const float*)d_in[1];
    const float* W_lin = (const float*)d_in[2];
    const float* b_lin = (const float*)d_in[3];
    const float* Wk    = (const float*)d_in[4];
    const float* bk    = (const float*)d_in[5];
    const float* Wq    = (const float*)d_in[6];
    const float* bq    = (const float*)d_in[7];
    const float* Wv    = (const float*)d_in[8];
    const float* bv    = (const float*)d_in[9];

    const size_t nPad = (size_t)B_ * S_ * DP_;   // f16 elements per padded tensor
    _Float16* Kp = (_Float16*)d_ws;
    _Float16* Qp = Kp + nPad;
    _Float16* Vt = Qp + nPad;

    // Projections: one thread per (b,s) row.
    proj_kernel<<<(B_ * S_ + 255) / 256, 256, 0, stream>>>(
        x1, x2, W_lin, b_lin, Wk, bk, Wq, bq, Wv, bv, Kp, Qp, Vt);

    // Flash attention: B*(S/16) = 2048 waves, 8 waves per block -> 256 blocks.
    attn_kernel<<<(B_ * (S_ / 16)) / 8, 256, 0, stream>>>(
        Kp, Qp, Vt, (float*)d_out);
}
